// MultiViewChunkRetrieval_51634096832665
// MI455X (gfx1250) — matrix operand order repaired and checked by reading, not compiled, not run on hardware
//
#include <hip/hip_runtime.h>

typedef __attribute__((ext_vector_type(8)))  __bf16 v8bf;
typedef __attribute__((ext_vector_type(16))) __bf16 v16bf;
typedef __attribute__((ext_vector_type(8)))  float  v8f;

#define TM 128
#define TN 64
#define TK 32
#define LSTR (TK + 8)

// ---------------- block reduction helpers (blockDim.x == 256) ----------------
__device__ __forceinline__ float blk_sum256(float v, float* red) {
  int t = threadIdx.x;
  red[t] = v; __syncthreads();
  for (int s = 128; s > 0; s >>= 1) { if (t < s) red[t] += red[t + s]; __syncthreads(); }
  float r = red[0]; __syncthreads();
  return r;
}
__device__ __forceinline__ float blk_max256(float v, float* red) {
  int t = threadIdx.x;
  red[t] = v; __syncthreads();
  for (int s = 128; s > 0; s >>= 1) { if (t < s) red[t] = fmaxf(red[t], red[t + s]); __syncthreads(); }
  float r = red[0]; __syncthreads();
  return r;
}

__device__ __forceinline__ v8bf cvt8(float4 a, float4 b) {
  v8bf o;
  o[0] = (__bf16)a.x; o[1] = (__bf16)a.y; o[2] = (__bf16)a.z; o[3] = (__bf16)a.w;
  o[4] = (__bf16)b.x; o[5] = (__bf16)b.y; o[6] = (__bf16)b.z; o[7] = (__bf16)b.w;
  return o;
}

// ---------------- generic f32->bf16 WMMA GEMM:  C = scale*(A @ BT^T) + bias --
// A: [M x Kd] f32 (lda), batch offset (z % a_zmod)*a_bs
// BT: [N x Kd] f32 (ldbt) = B transposed, batch offset (z % bt_zmod)*bt_bs
// C: [M x N] f32 (ldc), batch offset z*c_bs
// 128x64 tile, 8 waves, 4 WMMA/wave/iter; register-prefetch + ping-pong LDS,
// one barrier per K-tile.
__global__ __launch_bounds__(256) void wmma_gemm(
    const float* __restrict__ A, int lda, long a_bs, int a_zmod,
    const float* __restrict__ BT, int ldbt, long bt_bs, int bt_zmod,
    const float* __restrict__ bias,
    float* __restrict__ C, int ldc, long c_bs,
    int M, int Kd, float scale)
{
  const int bz = blockIdx.z;
  A  += (long)(bz % a_zmod) * a_bs;
  BT += (long)(bz % bt_zmod) * bt_bs;
  C  += (long)bz * c_bs;
  const int m0 = blockIdx.y * TM;
  const int n0 = blockIdx.x * TN;

  __shared__ __attribute__((aligned(16))) __bf16 lA[2][TM][LSTR];
  __shared__ __attribute__((aligned(16))) __bf16 lB[2][TN][LSTR];

  const int tid  = threadIdx.x;
  const int lane = tid & 31;
  const int wave = tid >> 5;       // 0..7 -> m sub-tile
  const int hl   = lane >> 4;      // 0/1
  const int l16  = lane & 15;

  v8f acc[4] = {v8f{}, v8f{}, v8f{}, v8f{}};

  const int rstage = tid >> 2;            // 0..63
  const int cstage = (tid & 3) * 8;       // 0,8,16,24

  const int  gm0     = m0 + rstage;            // A rows rstage and rstage+64
  const int  gm1     = gm0 + 64;
  const bool a0_ok   = (gm0 < M);
  const bool a1_ok   = (gm1 < M);
  const float* Abase0 = A + (long)gm0 * lda + cstage;
  const float* Abase1 = A + (long)gm1 * lda + cstage;
  const float* Bbase  = BT + (long)(n0 + rstage) * ldbt + cstage;

  // prefetch first K-tile into registers
  float4 ra0 = {}, ra1 = {}, ra2 = {}, ra3 = {}, rb0, rb1;
  if (a0_ok) {
    ra0 = ((const float4*)Abase0)[0];
    ra1 = ((const float4*)Abase0)[1];
  }
  if (a1_ok) {
    ra2 = ((const float4*)Abase1)[0];
    ra3 = ((const float4*)Abase1)[1];
  }
  rb0 = ((const float4*)Bbase)[0];
  rb1 = ((const float4*)Bbase)[1];

  int buf = 0;
  for (int k0 = 0; k0 < Kd; k0 += TK, buf ^= 1) {
    // commit prefetched tile to LDS (b128 stores)
    *(v8bf*)&lA[buf][rstage][cstage]      = cvt8(ra0, ra1);
    *(v8bf*)&lA[buf][rstage + 64][cstage] = cvt8(ra2, ra3);
    *(v8bf*)&lB[buf][rstage][cstage]      = cvt8(rb0, rb1);
    __syncthreads();

    // prefetch next tile while WMMA consumes this one
    const int kn = k0 + TK;
    if (kn < Kd) {
      if (a0_ok) {
        ra0 = ((const float4*)(Abase0 + kn))[0];
        ra1 = ((const float4*)(Abase0 + kn))[1];
      }
      if (a1_ok) {
        ra2 = ((const float4*)(Abase1 + kn))[0];
        ra3 = ((const float4*)(Abase1 + kn))[1];
      }
      rb0 = ((const float4*)(Bbase + kn))[0];
      rb1 = ((const float4*)(Bbase + kn))[1];
    }

    // A fragment: 16x32, lane(h,m): elems 0..7 -> k=h*8+i, 8..15 -> k=16+h*8+i
    const __bf16* arow = &lA[buf][wave * 16 + l16][0];
    v8bf a_lo = *(const v8bf*)(arow + hl * 8);
    v8bf a_hi = *(const v8bf*)(arow + 16 + hl * 8);
    v16bf af = __builtin_shufflevector(a_lo, a_hi,
        0,1,2,3,4,5,6,7,8,9,10,11,12,13,14,15);

    // B fragments: 32x16, lane(h,n): elems 0..7 -> k=h*8+i, 8..15 -> k=16+h*8+i  (BT row-major)
    #pragma unroll
    for (int t = 0; t < 4; ++t) {
      const __bf16* brow = &lB[buf][t * 16 + l16][0];
      v8bf b_lo = *(const v8bf*)(brow + hl * 8);
      v8bf b_hi = *(const v8bf*)(brow + 16 + hl * 8);
      v16bf bfr = __builtin_shufflevector(b_lo, b_hi,
          0,1,2,3,4,5,6,7,8,9,10,11,12,13,14,15);
      acc[t] = __builtin_amdgcn_wmma_f32_16x16x32_bf16(false, af, false, bfr,
                                                       (short)0, acc[t], false, false);
    }
  }

  // C layout: lane(h,n): VGPR r -> row m = h*8 + r, col n
  const int cm = m0 + wave * 16 + hl * 8;
  #pragma unroll
  for (int t = 0; t < 4; ++t) {
    const int cn = n0 + t * 16 + l16;
    const float bb = bias ? bias[cn] : 0.0f;
    #pragma unroll
    for (int r = 0; r < 8; ++r) {
      int gmr = cm + r;
      if (gmr < M) C[(long)gmr * ldc + cn] = acc[t][r] * scale + bb;
    }
  }
}

// ---------------- f32 transpose: dst[c][r] = src[r][c] -------------------
__global__ __launch_bounds__(256) void transpose_f32(
    const float* __restrict__ src, float* __restrict__ dst, int rows, int cols) {
  __shared__ float tile[32][33];
  int c0 = blockIdx.x * 32, r0 = blockIdx.y * 32;
  int tx = threadIdx.x & 31, ty = threadIdx.x >> 5;
  for (int i = ty; i < 32; i += 8)
    tile[i][tx] = src[(long)(r0 + i) * cols + (c0 + tx)];
  __syncthreads();
  for (int i = ty; i < 32; i += 8)
    dst[(long)(c0 + i) * rows + (r0 + tx)] = tile[tx][i];
}

// ---------------- [B*Q][H] -> [B][nh][Q][d] repack ------------------------
__global__ __launch_bounds__(256) void repack_heads(
    const float* __restrict__ qh, float* __restrict__ out,
    int B_, int Q_, int nh_, int d_) {
  long H_ = (long)nh_ * d_;
  long total = (long)B_ * Q_ * H_;
  long idx = (long)blockIdx.x * 256 + threadIdx.x;
  if (idx >= total) return;
  long c = idx % H_; long q = (idx / H_) % Q_; long b = idx / (H_ * Q_);
  long h = c / d_, dd = c % d_;
  out[(((b * nh_ + h) * Q_) + q) * d_ + dd] = qh[idx];
}

// ---------------- per-row max & sum(exp) over length Lrow -----------------
__global__ __launch_bounds__(256) void row_stats(
    const float* __restrict__ S, int Lrow, float* __restrict__ stats) {
  __shared__ float red[256];
  long row = blockIdx.x;
  const float* p = S + row * (long)Lrow;
  float m = -3.0e38f;
  for (int i = threadIdx.x; i < Lrow; i += 256) m = fmaxf(m, p[i]);
  m = blk_max256(m, red);
  float z = 0.0f;
  for (int i = threadIdx.x; i < Lrow; i += 256) z += __expf(p[i] - m);
  z = blk_sum256(z, red);
  if (threadIdx.x == 0) { stats[row * 2] = m; stats[row * 2 + 1] = z; }
}

// ---- global scorer chunk accumulation: mean_{hq,l} exp(s-m)/Z ------------
__global__ __launch_bounds__(256) void global_chunk(
    const float* __restrict__ S, const float* __restrict__ stats,
    float* __restrict__ out, int HQ, int Krow, int Lc, int J_) {
  __shared__ float red[256];
  int b = blockIdx.y, j = blockIdx.x;
  const float* Sb  = S + (long)b * HQ * Krow;
  const float* stb = stats + (long)b * HQ * 2;
  float acc = 0.0f;
  int tot = HQ * Lc;
  for (int idx = threadIdx.x; idx < tot; idx += 256) {
    int r = idx / Lc, l = idx - r * Lc;
    acc += __expf(Sb[(long)r * Krow + j * Lc + l] - stb[r * 2]) / stb[r * 2 + 1];
  }
  acc = blk_sum256(acc, red);
  if (threadIdx.x == 0) out[b * J_ + j] = acc / (float)tot;
}

// ---- local scorer: mean_{hq} ( max_l softmax_l ) = mean 1/Z_chunk --------
__global__ __launch_bounds__(256) void local_chunk(
    const float* __restrict__ S, float* __restrict__ out,
    int HQ, int Krow, int Lc, int J_) {
  __shared__ float red[256];
  int b = blockIdx.y, j = blockIdx.x;
  const float* Sb = S + (long)b * HQ * Krow + (long)j * Lc;
  float acc = 0.0f;
  for (int r = threadIdx.x; r < HQ; r += 256) {
    const float* p = Sb + (long)r * Krow;
    float m = -3.0e38f;
    for (int l = 0; l < Lc; ++l) m = fmaxf(m, p[l]);
    float z = 0.0f;
    for (int l = 0; l < Lc; ++l) z += __expf(p[l] - m);
    acc += 1.0f / z;
  }
  acc = blk_sum256(acc, red);
  if (threadIdx.x == 0) out[b * J_ + j] = acc / (float)HQ;
}

// ---------------- softmax over 64-wide rows -------------------------------
__global__ void row_softmax64(const float* __restrict__ in, float* __restrict__ out, int cols) {
  __shared__ float red[64];
  int row = blockIdx.x, t = threadIdx.x;
  float v = in[row * cols + t];
  red[t] = v; __syncthreads();
  for (int s = 32; s > 0; s >>= 1) { if (t < s) red[t] = fmaxf(red[t], red[t + s]); __syncthreads(); }
  float m = red[0]; __syncthreads();
  float e = __expf(v - m);
  red[t] = e; __syncthreads();
  for (int s = 32; s > 0; s >>= 1) { if (t < s) red[t] += red[t + s]; __syncthreads(); }
  out[row * cols + t] = e / red[0];
}

// ---------------- question mean-pool + LayerNorm --------------------------
__global__ __launch_bounds__(256) void qvec_ln(
    const float* __restrict__ qemb, const float* __restrict__ g,
    const float* __restrict__ beta, float* __restrict__ qvec, int Q_, int H_) {
  __shared__ float red[256];
  int b = blockIdx.x;
  const float* p = qemb + (long)b * Q_ * H_;
  float vals[4]; float lsum = 0.0f;
  #pragma unroll
  for (int u = 0; u < 4; ++u) {
    int c = threadIdx.x + u * 256;
    float s = 0.0f;
    for (int q = 0; q < Q_; ++q) s += p[(long)q * H_ + c];
    vals[u] = s / (float)Q_;
    lsum += vals[u];
  }
  float mu = blk_sum256(lsum, red) / (float)H_;
  float lv = 0.0f;
  #pragma unroll
  for (int u = 0; u < 4; ++u) { float d = vals[u] - mu; lv += d * d; }
  float var = blk_sum256(lv, red) / (float)H_;
  float rs = rsqrtf(var + 1e-5f);
  #pragma unroll
  for (int u = 0; u < 4; ++u) {
    int c = threadIdx.x + u * 256;
    qvec[(long)b * H_ + c] = (vals[u] - mu) * rs * g[c] + beta[c];
  }
}

// -------- chunk mean-pool + LayerNorm + build g_in = [qvec | cvec] --------
__global__ __launch_bounds__(256) void cvec_ln_gin(
    const float* __restrict__ ctx, const float* __restrict__ qvec,
    const float* __restrict__ g, const float* __restrict__ beta,
    float* __restrict__ gin, int Lr, int H_, int J_) {
  __shared__ float red[256];
  int b = blockIdx.y, j = blockIdx.x;
  const float* p = ctx + ((long)b * J_ + j) * Lr * H_;
  float vals[4]; float lsum = 0.0f;
  #pragma unroll
  for (int u = 0; u < 4; ++u) {
    int c = threadIdx.x + u * 256;
    float s = 0.0f;
    for (int l = 0; l < Lr; ++l) s += p[(long)l * H_ + c];
    vals[u] = s / (float)Lr;
    lsum += vals[u];
  }
  float mu = blk_sum256(lsum, red) / (float)H_;
  float lv = 0.0f;
  #pragma unroll
  for (int u = 0; u < 4; ++u) { float d = vals[u] - mu; lv += d * d; }
  float var = blk_sum256(lv, red) / (float)H_;
  float rs = rsqrtf(var + 1e-5f);
  long bj = (long)b * J_ + j;
  #pragma unroll
  for (int u = 0; u < 4; ++u) {
    int c = threadIdx.x + u * 256;
    gin[bj * 2 * H_ + H_ + c] = (vals[u] - mu) * rs * g[c] + beta[c];
    gin[bj * 2 * H_ + c]      = qvec[(long)b * H_ + c];
  }
}

// ---- gate = sigmoid(gelu(h1) . W2 + b2); fused = g*loc + (1-g)*glob ------
__global__ __launch_bounds__(256) void gate_fuse(
    const float* __restrict__ h1, const float* __restrict__ W2,
    const float* __restrict__ b2, const float* __restrict__ locS,
    const float* __restrict__ globS, float* __restrict__ fused, int H_) {
  __shared__ float red[256];
  int bj = blockIdx.x;
  const float* p = h1 + (long)bj * H_;
  float s = 0.0f;
  for (int c = threadIdx.x; c < H_; c += 256) {
    float x = p[c];
    float ge = 0.5f * x * (1.0f + erff(x * 0.70710678118f));  // exact gelu
    s += ge * W2[c];
  }
  s = blk_sum256(s, red);
  if (threadIdx.x == 0) {
    float gt = 1.0f / (1.0f + __expf(-(s + b2[0])));
    fused[bj] = gt * locS[bj] + (1.0f - gt) * globS[bj];
  }
}

// ---------------- final softmax over J + gate mean ------------------------
__global__ void final_softmax64(const float* __restrict__ fused,
                                float* __restrict__ outF, float* __restrict__ outG,
                                int cols) {
  __shared__ float red[64];
  int b = blockIdx.x, t = threadIdx.x;
  float v = fused[b * cols + t];
  red[t] = v; __syncthreads();
  for (int s = 32; s > 0; s >>= 1) { if (t < s) red[t] = fmaxf(red[t], red[t + s]); __syncthreads(); }
  float m = red[0]; __syncthreads();
  float e = __expf(v - m);
  red[t] = e; __syncthreads();
  for (int s = 32; s > 0; s >>= 1) { if (t < s) red[t] += red[t + s]; __syncthreads(); }
  float Z = red[0]; __syncthreads();
  float sm = e / Z;
  outF[b * cols + t] = sm;
  red[t] = sm; __syncthreads();
  for (int s = 32; s > 0; s >>= 1) { if (t < s) red[t] += red[t + s]; __syncthreads(); }
  if (t == 0) outG[b] = red[0] / (float)cols;
}

// ======================== host orchestration ==============================
extern "C" void kernel_launch(void* const* d_in, const int* in_sizes, int n_in,
                              void* d_out, int out_size, void* d_ws, size_t ws_size,
                              hipStream_t stream) {
  constexpr int B = 4, Q = 32, K = 8192, H = 1024, NH = 16, Lc = 128;
  constexpr int J = K / Lc;           // 64
  constexpr int D = H / NH;           // 64
  constexpr int HQ = NH * Q;          // 512
  const float scale = 0.125f;         // 1/sqrt(64)

  const float* question = (const float*)d_in[0];
  const float* ctx      = (const float*)d_in[1];
  const float* Wq_l = (const float*)d_in[2];
  const float* bq_l = (const float*)d_in[3];
  const float* Wk_l = (const float*)d_in[4];   // bk_l cancels in softmax
  const float* Wq_g = (const float*)d_in[6];
  const float* bq_g = (const float*)d_in[7];
  const float* Wk_g = (const float*)d_in[8];   // bk_g cancels in softmax
  const float* ln_g = (const float*)d_in[10];
  const float* ln_b = (const float*)d_in[11];
  const float* W1   = (const float*)d_in[12];
  const float* b1   = (const float*)d_in[13];
  const float* W2   = (const float*)d_in[14];
  const float* b2   = (const float*)d_in[15];
  (void)in_sizes; (void)n_in; (void)out_size; (void)ws_size;

  float* out = (float*)d_out;  // [final(256) | local(256) | global(256) | gate(4)]
  char*  ws  = (char*)d_ws;

  size_t off = 0;
  auto A = [&](size_t bytes) { size_t o = off; off += (bytes + 255) & ~(size_t)255; return o; };
  size_t o_wqgT = A((size_t)H * H * 4);
  size_t o_wqlT = A((size_t)H * H * 4);
  size_t o_w1T  = A((size_t)H * 2 * H * 4);
  size_t o_qh   = A((size_t)B * Q * H * 4);          // scratch, reused g then l
  size_t o_qhgh = A((size_t)B * Q * H * 4);          // head-major global
  size_t o_qhlh = A((size_t)B * Q * H * 4);          // head-major local
  size_t o_P    = A((size_t)B * HQ * H * 4);         // 8 MB, reused
  size_t o_S    = A((size_t)B * HQ * K * 4);         // 64 MB, reused
  size_t o_st   = A((size_t)B * HQ * 2 * 4);
  size_t o_gsc  = A((size_t)B * J * 4);
  size_t o_lsc  = A((size_t)B * J * 4);
  size_t o_qv   = A((size_t)B * H * 4);
  size_t o_gin  = A((size_t)B * J * 2 * H * 4);
  size_t o_h1   = A((size_t)B * J * H * 4);
  size_t o_fu   = A((size_t)B * J * 4);

  float* wqgT = (float*)(ws + o_wqgT);
  float* wqlT = (float*)(ws + o_wqlT);
  float* w1T  = (float*)(ws + o_w1T);
  float* qh   = (float*)(ws + o_qh);
  float* qhgh = (float*)(ws + o_qhgh);
  float* qhlh = (float*)(ws + o_qhlh);
  float* P    = (float*)(ws + o_P);
  float* S    = (float*)(ws + o_S);
  float* st   = (float*)(ws + o_st);
  float* gsc  = (float*)(ws + o_gsc);
  float* lsc  = (float*)(ws + o_lsc);
  float* qv   = (float*)(ws + o_qv);
  float* gin  = (float*)(ws + o_gin);
  float* h1   = (float*)(ws + o_h1);
  float* fu   = (float*)(ws + o_fu);

  // 1. transposes (Wq for A@W^T form; W1 for router GEMM)
  transpose_f32<<<dim3(H / 32, H / 32), 256, 0, stream>>>(Wq_g, wqgT, H, H);
  transpose_f32<<<dim3(H / 32, H / 32), 256, 0, stream>>>(Wq_l, wqlT, H, H);
  transpose_f32<<<dim3(H / 32, 2 * H / 32), 256, 0, stream>>>(W1, w1T, 2 * H, H);

  // 2. global path: q-proj -> repack -> P -> scores -> stats -> chunk -> softmax
  wmma_gemm<<<dim3(H / TN, (B * Q + TM - 1) / TM, 1), 256, 0, stream>>>(
      question, H, 0, 1, wqgT, H, 0, 1, bq_g, qh, H, 0, B * Q, H, 1.0f);
  repack_heads<<<(B * Q * H + 255) / 256, 256, 0, stream>>>(qh, qhgh, B, Q, NH, D);
  wmma_gemm<<<dim3(H / TN, 1, B * NH), 256, 0, stream>>>(
      qhgh, D, (long)Q * D, B * NH, Wk_g, H, (long)D, NH, nullptr,
      P, H, (long)Q * H, Q, D, 1.0f);
  wmma_gemm<<<dim3(K / TN, HQ / TM, B), 256, 0, stream>>>(
      P, H, (long)HQ * H, B, ctx, H, (long)K * H, B, nullptr,
      S, K, (long)HQ * K, HQ, H, scale);
  row_stats<<<B * HQ, 256, 0, stream>>>(S, K, st);
  global_chunk<<<dim3(J, B), 256, 0, stream>>>(S, st, gsc, HQ, K, Lc, J);
  row_softmax64<<<B, 64, 0, stream>>>(gsc, out + 2 * B * J, J);   // global_scores

  // 3. local path (reuses P and S buffers)
  wmma_gemm<<<dim3(H / TN, (B * Q + TM - 1) / TM, 1), 256, 0, stream>>>(
      question, H, 0, 1, wqlT, H, 0, 1, bq_l, qh, H, 0, B * Q, H, 1.0f);
  repack_heads<<<(B * Q * H + 255) / 256, 256, 0, stream>>>(qh, qhlh, B, Q, NH, D);
  wmma_gemm<<<dim3(H / TN, 1, B * NH), 256, 0, stream>>>(
      qhlh, D, (long)Q * D, B * NH, Wk_l, H, (long)D, NH, nullptr,
      P, H, (long)Q * H, Q, D, 1.0f);
  wmma_gemm<<<dim3(K / TN, HQ / TM, B), 256, 0, stream>>>(
      P, H, (long)HQ * H, B, ctx, H, (long)K * H, B, nullptr,
      S, K, (long)HQ * K, HQ, H, scale);
  local_chunk<<<dim3(J, B), 256, 0, stream>>>(S, lsc, HQ, K, Lc, J);
  row_softmax64<<<B, 64, 0, stream>>>(lsc, out + B * J, J);       // local_scores

  // 4. router
  qvec_ln<<<B, 256, 0, stream>>>(question, ln_g, ln_b, qv, Q, H);
  cvec_ln_gin<<<dim3(J, B), 256, 0, stream>>>(ctx, qv, ln_g, ln_b, gin, Lc, H, J);
  wmma_gemm<<<dim3(H / TN, (B * J + TM - 1) / TM, 1), 256, 0, stream>>>(
      gin, 2 * H, 0, 1, w1T, 2 * H, 0, 1, b1, h1, H, 0, B * J, 2 * H, 1.0f);
  gate_fuse<<<B * J, 256, 0, stream>>>(h1, W2, b2, out + B * J, out + 2 * B * J, fu, H);
  final_softmax64<<<B, 64, 0, stream>>>(fu, out, out + 3 * B * J, J);
}
